// CrossGroupedQueryAttention_69088843924231
// MI455X (gfx1250) — matrix operand order repaired and checked by reading, not compiled, not run on hardware
//
#include <hip/hip_runtime.h>
#include <hip/hip_fp16.h>

// ---------------------------------------------------------------------------
// CrossGroupedQueryAttention for MI455X (gfx1250, wave32, WMMA f16->f32)
// Pipeline: [gemm Q/K/V] -> [rope+pack f16] -> [flash attention] -> [gemm out]
// All matrix work through v_wmma_f32_16x16x32_f16; tiles staged via LDS with
// global_prefetch_b8 ahead of each stage; mask fast-path skips clean tiles.
// ---------------------------------------------------------------------------

typedef __attribute__((ext_vector_type(16))) _Float16 v16h;
typedef __attribute__((ext_vector_type(8)))  _Float16 v8h;
typedef __attribute__((ext_vector_type(4)))  _Float16 v4h;
typedef __attribute__((ext_vector_type(8)))  float    v8f;

#define B_     2
#define SQ_    2048
#define SKV_   2048
#define EMB_   1024
#define H_     16
#define G_     4
#define DH_    64
#define HDTOT_ 1024   // H_*DH_

__device__ __forceinline__ v16h cat16(v8h lo, v8h hi) {
  return __builtin_shufflevector(lo, hi, 0,1,2,3,4,5,6,7,8,9,10,11,12,13,14,15);
}
__device__ __forceinline__ v8f wmma16x16x32(v16h a, v16h b, v8f c) {
  // D = A(16x32 f16) x B(32x16 f16) + C(16x16 f32)
  return __builtin_amdgcn_wmma_f32_16x16x32_f16(false, a, false, b, (short)0, c,
                                                false, false);
}

// ---------------------------------------------------------------------------
// Generic C = A * W^T   (A: [M][K] f32 or f16, W: [N][K] f32, C: [M][N] f32)
// Block tile 128(M) x 64(N), staged K-chunk 64 (two 32-wide WMMA sub-steps).
// 256 threads = 8 waves (4 M x 2 N); each wave computes a 32x32 patch as
// 2x2 WMMA fragments. Next tiles prefetched before the staging barrier.
// ---------------------------------------------------------------------------
template<bool A_F16>
__global__ __launch_bounds__(256)
void gemm_xwt_kernel(const void* __restrict__ Ap, const float* __restrict__ W,
                     float* __restrict__ C, int M, int N, int K)
{
  constexpr int MB = 128, NB = 64, KB = 64;
  constexpr int LDA = KB + 8, LDB = KB + 8;      // padded rows (16B aligned)
  __shared__ __align__(16) _Float16 As[MB * LDA];
  __shared__ __align__(16) _Float16 Bs[NB * LDB];

  const int tid  = threadIdx.x;
  const int lane = tid & 31;
  const int wave = tid >> 5;
  const int half = lane >> 4;
  const int l16  = lane & 15;
  const int wm   = wave >> 1;                    // 0..3
  const int wn   = wave & 1;                     // 0..1
  const int m0   = blockIdx.x * MB;
  const int n0   = blockIdx.y * NB;

  v8f acc[2][2] = {};

  for (int k0 = 0; k0 < K; k0 += KB) {
    // prefetch next K-chunk into cache while this stage computes
    if (k0 + KB < K) {
      const int nk = k0 + KB;
      const size_t aesz = A_F16 ? 2 : 4;
      const char* Ab = (const char*)Ap;
      __builtin_prefetch(Ab + ((size_t)(m0 + (tid & 127)) * K + nk) * aesz +
                         (tid >> 7) * 128, 0, 0);
      __builtin_prefetch((const char*)W +
                         ((size_t)(n0 + (tid & 63)) * K + nk) * 4 +
                         ((tid >> 6) & 3) * 64, 0, 0);
    }
    __syncthreads();
    if (A_F16) {
      const _Float16* A = (const _Float16*)Ap;
      #pragma unroll
      for (int i = 0; i < 4; ++i) {              // 1024 x v8h, 8 per row
        int idx = tid + i * 256;
        int row = idx >> 3;
        int kc  = (idx & 7) * 8;
        *(v8h*)(As + row * LDA + kc) =
            *(const v8h*)(A + (size_t)(m0 + row) * K + k0 + kc);
      }
    } else {
      const float* A = (const float*)Ap;
      #pragma unroll
      for (int i = 0; i < 8; ++i) {              // 2048 x float4, 16 per row
        int idx = tid + i * 256;
        int row = idx >> 4;
        int kc  = (idx & 15) * 4;
        float4 v = *(const float4*)(A + (size_t)(m0 + row) * K + k0 + kc);
        v4h h = {(_Float16)v.x, (_Float16)v.y, (_Float16)v.z, (_Float16)v.w};
        *(v4h*)(As + row * LDA + kc) = h;
      }
    }
    #pragma unroll
    for (int i = 0; i < 4; ++i) {                // 1024 x float4, 16 per row
      int idx = tid + i * 256;
      int row = idx >> 4;
      int kc  = (idx & 15) * 4;
      float4 v = *(const float4*)(W + (size_t)(n0 + row) * K + k0 + kc);
      v4h h = {(_Float16)v.x, (_Float16)v.y, (_Float16)v.z, (_Float16)v.w};
      *(v4h*)(Bs + row * LDB + kc) = h;
    }
    __syncthreads();

    #pragma unroll
    for (int ks = 0; ks < 2; ++ks) {             // two 32-wide WMMA sub-steps
      const int ko = ks * 32;
      v16h af[2], bf[2];
      #pragma unroll
      for (int mi = 0; mi < 2; ++mi) {
        const _Float16* p = As + (wm * 32 + mi * 16 + l16) * LDA + ko;
        af[mi] = cat16(*(const v8h*)(p + 8 * half),
                       *(const v8h*)(p + 16 + 8 * half));
      }
      #pragma unroll
      for (int ni = 0; ni < 2; ++ni) {
        const _Float16* p = Bs + (wn * 32 + ni * 16 + l16) * LDB + ko;
        bf[ni] = cat16(*(const v8h*)(p + 8 * half),
                       *(const v8h*)(p + 16 + 8 * half));
      }
      #pragma unroll
      for (int mi = 0; mi < 2; ++mi)
        #pragma unroll
        for (int ni = 0; ni < 2; ++ni)
          acc[mi][ni] = wmma16x16x32(af[mi], bf[ni], acc[mi][ni]);
    }
  }

  #pragma unroll
  for (int mi = 0; mi < 2; ++mi)
    #pragma unroll
    for (int ni = 0; ni < 2; ++ni)
      #pragma unroll
      for (int r = 0; r < 8; ++r) {
        int m = m0 + wm * 32 + mi * 16 + r + 8 * half;
        int n = n0 + wn * 32 + ni * 16 + l16;
        C[(size_t)m * N + n] = acc[mi][ni][r];
      }
}

// ---------------------------------------------------------------------------
// RoPE + f32 -> f16 pack:  X [B*S][nh*64]  ->  Y [B][nh][S][64]
// One thread per (b,h,s,d<32); handles the (d, d+32) rotation pair.
// ---------------------------------------------------------------------------
__global__ __launch_bounds__(256)
void rope_pack_kernel(const float* __restrict__ X, _Float16* __restrict__ Y,
                      const float* __restrict__ cs, const float* __restrict__ sn,
                      int S, int nh, int do_rope)
{
  int idx = blockIdx.x * blockDim.x + threadIdx.x;   // B_*nh*S*32 threads exactly
  int d = idx & 31;
  int s = (idx >> 5) % S;
  int h = ((idx >> 5) / S) % nh;
  int b = idx / (32 * S * nh);

  const float* xp = X + (size_t)(b * S + s) * (nh * 64) + h * 64 + d;
  float x1 = xp[0], x2 = xp[32];
  float o1, o2;
  if (do_rope) {
    float c1 = cs[s * 64 + d],      s1 = sn[s * 64 + d];
    float c2 = cs[s * 64 + d + 32], s2 = sn[s * 64 + d + 32];
    o1 = x1 * c1 - x2 * s1;          // rot[d]    = -x[d+32]
    o2 = x2 * c2 + x1 * s2;          // rot[d+32] =  x[d]
  } else { o1 = x1; o2 = x2; }

  _Float16* yp = Y + (((size_t)b * nh + h) * S + s) * 64 + d;
  yp[0]  = (_Float16)o1;
  yp[32] = (_Float16)o2;
}

// ---------------------------------------------------------------------------
// Flash attention: 4 waves/block, each wave owns 16 q rows; block covers 64 q
// rows of one (b,h). 64-wide K/V tiles staged in LDS (V stored transposed via
// paired-b32 stores), two 32-kv WMMA sub-steps per stage. Online softmax with
// 16-lane shfl reductions. Mask tiles checked wave-uniformly: clean tiles
// (the common case) skip all per-element mask work.
// ---------------------------------------------------------------------------
__global__ __launch_bounds__(128)
void attn_kernel(const _Float16* __restrict__ Qh,   // [B][H][S][64]
                 const _Float16* __restrict__ Kh,   // [B][G][S][64]
                 const _Float16* __restrict__ Vh,   // [B][G][S][64]
                 const unsigned char* __restrict__ mask, // [SQ][SKV]
                 _Float16* __restrict__ O)          // [B*SQ][H*64]
{
  constexpr int KT  = 64;          // kv tile
  constexpr int LDK = DH_ + 8;     // 72 f16 per K row
  constexpr int LDV = KT + 8;      // 72 f16 per V^T row
  __shared__ __align__(16) _Float16 Ks[KT * LDK];
  __shared__ __align__(16) _Float16 Vt[DH_ * LDV];
  __shared__ __align__(16) _Float16 Ps[4 * 16 * 32];       // per-wave P scratch
  __shared__ __align__(16) unsigned char Ms[4 * 16 * KT];  // per-wave mask tile

  const int tid  = threadIdx.x;
  const int lane = tid & 31, wave = tid >> 5;
  const int half = lane >> 4, l16 = lane & 15;
  const int h = blockIdx.y, b = blockIdx.z;
  const int g = h >> 2;                         // rep = H/G = 4
  const int qbase = blockIdx.x * 64 + wave * 16;

  // Q fragments: 16 q rows x 64 d, two 16x32 A-frags, loaded once
  const _Float16* Qp = Qh + (((size_t)b * H_ + h) * SQ_ + qbase + l16) * DH_;
  v16h qf[2];
  #pragma unroll
  for (int d = 0; d < 2; ++d)
    qf[d] = cat16(*(const v8h*)(Qp + d * 32 + 8 * half),
                  *(const v8h*)(Qp + d * 32 + 16 + 8 * half));

  float mr[8], lr[8], ar[8];
  #pragma unroll
  for (int r = 0; r < 8; ++r) { mr[r] = -1e30f; lr[r] = 0.f; }
  v8f o[4] = {};

  const _Float16* Kp0 = Kh + ((size_t)b * G_ + g) * SKV_ * DH_;
  const _Float16* Vp0 = Vh + ((size_t)b * G_ + g) * SKV_ * DH_;

  for (int kv0 = 0; kv0 < SKV_; kv0 += KT) {
    // prefetch next tile (one 128B row per thread: K for tid<64, V for tid>=64)
    if (kv0 + KT < SKV_) {
      const _Float16* pb = (tid < 64) ? Kp0 : Vp0;
      __builtin_prefetch(pb + (size_t)(kv0 + KT + (tid & 63)) * DH_, 0, 0);
    }
    __syncthreads();
    // K tile -> LDS [kv][d]
    #pragma unroll
    for (int i = 0; i < 4; ++i) {
      int idx = tid + i * 128;                   // 512 x v8h, 8 per row
      int row = idx >> 3, c = (idx & 7) * 8;
      *(v8h*)(Ks + row * LDK + c) =
          *(const v8h*)(Kp0 + (size_t)(kv0 + row) * DH_ + c);
    }
    // V tile -> LDS transposed [d][kv]: (2 kv x 4 d) units, paired-b32 stores
    #pragma unroll
    for (int i = 0; i < 4; ++i) {
      int u   = tid + i * 128;                   // 512 units
      int kvp = u >> 4;                          // 0..31 (kv pair)
      int dq  = (u & 15) * 4;                    // 0..60
      const _Float16* vp = Vp0 + (size_t)(kv0 + 2 * kvp) * DH_ + dq;
      v4h a = *(const v4h*)(vp);
      v4h c = *(const v4h*)(vp + DH_);
      #pragma unroll
      for (int j = 0; j < 4; ++j) {
        union { _Float16 hh[2]; unsigned uu; } pk;
        pk.hh[0] = a[j]; pk.hh[1] = c[j];
        *(unsigned*)(Vt + (dq + j) * LDV + 2 * kvp) = pk.uu;
      }
    }
    // per-wave mask tile (16 q x 64 kv), each lane grabs two 16B chunks
    uint4 mA = *(const uint4*)(mask + (size_t)(qbase + l16) * SKV_ + kv0 +
                               half * 16);
    uint4 mB = *(const uint4*)(mask + (size_t)(qbase + l16) * SKV_ + kv0 +
                               half * 16 + 32);
    *(uint4*)(Ms + (wave * 16 + l16) * KT + half * 16)      = mA;
    *(uint4*)(Ms + (wave * 16 + l16) * KT + half * 16 + 32) = mB;
    const bool dirty =
        __any((mA.x | mA.y | mA.z | mA.w | mB.x | mB.y | mB.z | mB.w) != 0);
    __syncthreads();

    #pragma unroll
    for (int sub = 0; sub < 2; ++sub) {          // two 32-kv sub-steps
      const int kvoff = sub * 32;

      // S = Q K^T : two 16x16 tiles, 2 k-steps over head dim
      v8f s[2] = {};
      #pragma unroll
      for (int t = 0; t < 2; ++t) {
        const _Float16* kp = Ks + (kvoff + t * 16 + l16) * LDK;
        #pragma unroll
        for (int d = 0; d < 2; ++d) {
          v16h bf = cat16(*(const v8h*)(kp + d * 32 + 8 * half),
                          *(const v8h*)(kp + d * 32 + 16 + 8 * half));
          s[t] = wmma16x16x32(qf[d], bf, s[t]);
        }
      }

      // scale, (sparse) mask, online softmax; row = r+8*half, col = t*16+l16
      #pragma unroll
      for (int r = 0; r < 8; ++r) {
        float s0 = s[0][r] * 0.125f;
        float s1 = s[1][r] * 0.125f;
        if (dirty) {
          const unsigned char* mrow =
              Ms + (wave * 16 + r + 8 * half) * KT + kvoff;
          if (mrow[l16])      s0 = -__builtin_inff();
          if (mrow[16 + l16]) s1 = -__builtin_inff();
        }
        float v = fmaxf(s0, s1);
        #pragma unroll
        for (int msk = 1; msk < 16; msk <<= 1)
          v = fmaxf(v, __shfl_xor(v, msk, 32));
        float mnew  = fmaxf(mr[r], v);
        float alpha = __expf(mr[r] - mnew);
        float p0 = __expf(s0 - mnew);
        float p1 = __expf(s1 - mnew);
        float ps = p0 + p1;
        #pragma unroll
        for (int msk = 1; msk < 16; msk <<= 1) ps += __shfl_xor(ps, msk, 32);
        lr[r] = lr[r] * alpha + ps;
        mr[r] = mnew;
        ar[r] = alpha;
        int prow = (wave * 16 + r + 8 * half) * 32;
        Ps[prow + l16]      = (_Float16)p0;
        Ps[prow + 16 + l16] = (_Float16)p1;
      }
      #pragma unroll
      for (int dt = 0; dt < 4; ++dt)
        #pragma unroll
        for (int r = 0; r < 8; ++r) o[dt][r] *= ar[r];

      // wave-internal cross-lane LDS dependency (C/D layout -> A layout)
      asm volatile("s_wait_dscnt 0" ::: "memory");

      // P A-fragment (16 q x 32 kv) and O += P V
      const _Float16* pp = Ps + (wave * 16 + l16) * 32;
      v16h pf = cat16(*(const v8h*)(pp + 8 * half),
                      *(const v8h*)(pp + 16 + 8 * half));
      #pragma unroll
      for (int dt = 0; dt < 4; ++dt) {
        const _Float16* vp = Vt + (dt * 16 + l16) * LDV + kvoff;
        v16h vf = cat16(*(const v8h*)(vp + 8 * half),
                        *(const v8h*)(vp + 16 + 8 * half));
        o[dt] = wmma16x16x32(pf, vf, o[dt]);
      }
    }
  }

  // epilogue: normalize and emit O as f16 rows [b*SQ+q][h*64+d]
  #pragma unroll
  for (int r = 0; r < 8; ++r) ar[r] = 1.0f / lr[r];
  #pragma unroll
  for (int dt = 0; dt < 4; ++dt)
    #pragma unroll
    for (int r = 0; r < 8; ++r) {
      int q   = qbase + r + 8 * half;
      int col = h * DH_ + dt * 16 + l16;
      O[(size_t)(b * SQ_ + q) * HDTOT_ + col] = (_Float16)(o[dt][r] * ar[r]);
    }
}

// ---------------------------------------------------------------------------
extern "C" void kernel_launch(void* const* d_in, const int* in_sizes, int n_in,
                              void* d_out, int out_size, void* d_ws, size_t ws_size,
                              hipStream_t stream)
{
  (void)in_sizes; (void)n_in; (void)out_size; (void)ws_size;

  const float* x_q  = (const float*)d_in[0];
  const float* x_kv = (const float*)d_in[1];
  const unsigned char* mask = (const unsigned char*)d_in[2];
  const float* cosb = (const float*)d_in[3];
  const float* sinb = (const float*)d_in[4];
  const float* Wq   = (const float*)d_in[5];
  const float* Wk   = (const float*)d_in[6];
  const float* Wv   = (const float*)d_in[7];
  const float* Wout = (const float*)d_in[8];
  float* out = (float*)d_out;

  // workspace layout (44 MiB total)
  char* w = (char*)d_ws;
  float*    Qf = (float*)(w);                        // 4096x1024 f32 = 16 MiB
  float*    Kf = (float*)(w + (16u << 20));          // 4096x256  f32 =  4 MiB
  float*    Vf = (float*)(w + (20u << 20));          // 4096x256  f32 =  4 MiB
  _Float16* Qh = (_Float16*)(w + (24u << 20));       // [B][H][S][64]  =  8 MiB
  _Float16* Kh = (_Float16*)(w + (32u << 20));       // [B][G][S][64]  =  2 MiB
  _Float16* Vh = (_Float16*)(w + (34u << 20));       // [B][G][S][64]  =  2 MiB
  _Float16* Oh = (_Float16*)(w + (36u << 20));       // 4096x1024 f16  =  8 MiB

  const int Mr = B_ * SQ_;  // 4096 rows

  // Q/K/V projections (f32 in, f32 out)
  gemm_xwt_kernel<false><<<dim3(Mr / 128, EMB_ / 64), 256, 0, stream>>>(
      x_q, Wq, Qf, Mr, HDTOT_, EMB_);
  gemm_xwt_kernel<false><<<dim3(Mr / 128, (G_ * DH_) / 64), 256, 0, stream>>>(
      x_kv, Wk, Kf, Mr, G_ * DH_, EMB_);
  gemm_xwt_kernel<false><<<dim3(Mr / 128, (G_ * DH_) / 64), 256, 0, stream>>>(
      x_kv, Wv, Vf, Mr, G_ * DH_, EMB_);

  // RoPE + pack to f16 (V: pack only)
  rope_pack_kernel<<<(B_ * H_ * SQ_ * 32) / 256, 256, 0, stream>>>(
      Qf, Qh, cosb, sinb, SQ_, H_, 1);
  rope_pack_kernel<<<(B_ * G_ * SKV_ * 32) / 256, 256, 0, stream>>>(
      Kf, Kh, cosb, sinb, SKV_, G_, 1);
  rope_pack_kernel<<<(B_ * G_ * SKV_ * 32) / 256, 256, 0, stream>>>(
      Vf, Vh, cosb, sinb, SKV_, G_, 0);

  // flash attention
  attn_kernel<<<dim3(SQ_ / 64, H_, B_), 128, 0, stream>>>(Qh, Kh, Vh, mask, Oh);

  // output projection (f16 A, f32 out)
  gemm_xwt_kernel<true><<<dim3(Mr / 128, HDTOT_ / 64), 256, 0, stream>>>(
      Oh, Wout, out, Mr, HDTOT_, EMB_);
}
